// SequenceGeneratorModel_38740605010268
// MI455X (gfx1250) — compile-verified, matrix operand order checked by reading, not run on hardware
//
#include <hip/hip_runtime.h>
#include <hip/hip_bf16.h>

#define VOCAB   50257
#define VPAD    50304   /* multiple of 64, pad cols zero-filled in weights */
#define DIM     1024
#define BATCH   32
#define NBEAM   4
#define NROWS   128     /* BATCH*NBEAM */
#define MAXLEN  32
#define BOS_T   0
#define EOS_T   2
#define PAD_T   1
#define NEG_BIG -3.0e38f

typedef __attribute__((ext_vector_type(16))) __bf16 v16bf;
typedef __attribute__((ext_vector_type(8)))  float  v8f;
typedef __hip_bfloat16 bf16_t;

// ---------------------------------------------------------------------------
// WMMA helpers. 16-bit A-matrix 16x32 layout (ISA 7.12.2): lane L(<16) holds
// row L, K = {0..7} in v0-3 and {16..23} in v4-7; lanes 16-31 hold K {8..15}
// and {24..31}.  Two contiguous 16B loads per lane.  B uses the same pattern
// on a pre-transposed ([N][K] row-major) weight copy.
// ---------------------------------------------------------------------------
__device__ __forceinline__ v16bf load_frag(const bf16_t* base, int stride, int lane) {
    int r  = lane & 15;
    int kb = (lane >> 4) << 3;          // 0 or 8
    const bf16_t* p = base + (long long)r * stride + kb;
    union { v16bf v; float4 f[2]; } u;
    u.f[0] = *reinterpret_cast<const float4*>(p);        // K = kb .. kb+7
    u.f[1] = *reinterpret_cast<const float4*>(p + 16);   // K = kb+16 .. kb+23
    return u.v;
}

__device__ __forceinline__ v8f wmma_bf16(v16bf a, v16bf b, v8f c) {
    return __builtin_amdgcn_wmma_f32_16x16x32_bf16(
        false, a, false, b, (short)0, c, false, false);
}

// 16x64-tile GEMM body: accumulates A[16xK] * B[Kx64] over K=DIM.
// Single fragment set per iteration: ~80 live VGPRs -> high wave occupancy,
// which is what hides load latency here (no accumulator spills).
__device__ __forceinline__ void gemm_16x64(const bf16_t* __restrict__ A,
                                           const bf16_t* __restrict__ Wt,
                                           int mbase, int nbase, int lane,
                                           v8f acc[4]) {
    const bf16_t* Ab = A + (long long)mbase * DIM;
    for (int k0 = 0; k0 < DIM; k0 += 32) {
        v16bf a = load_frag(Ab + k0, DIM, lane);
        v16bf b[4];
        #pragma unroll
        for (int j = 0; j < 4; ++j)
            b[j] = load_frag(Wt + (long long)(nbase + j * 16) * DIM + k0, DIM, lane);
        #pragma unroll
        for (int j = 0; j < 4; ++j) acc[j] = wmma_bf16(a, b[j], acc[j]);
    }
}

// ---------------------------------------------------------------------------
// One-time weight prep: dst[v][k] = bf16(src[k][v]), v padded with zeros.
// ---------------------------------------------------------------------------
__global__ void k_conv_T(const float* __restrict__ src, bf16_t* __restrict__ dst,
                         int srcR, int srcC, int dstRpad) {
    long long total = (long long)dstRpad * srcR;
    for (long long i = (long long)blockIdx.x * blockDim.x + threadIdx.x; i < total;
         i += (long long)gridDim.x * blockDim.x) {
        int k = (int)(i % srcR);
        int v = (int)(i / srcR);
        float x = (v < srcC) ? src[(long long)k * srcC + v] : 0.f;
        dst[i] = __float2bfloat16(x);
    }
}

__global__ void k_conv(const float* __restrict__ src, bf16_t* __restrict__ dst, int n) {
    for (int i = blockIdx.x * blockDim.x + threadIdx.x; i < n;
         i += gridDim.x * blockDim.x)
        dst[i] = __float2bfloat16(src[i]);
}

// ---------------------------------------------------------------------------
// h' = tanh(h @ w_h + emb[last]) ; writes fp32 (for reorder) and bf16 (for
// the logits GEMM).  One wave per 16x64 tile, 32 WMMAs over K=1024.
// ---------------------------------------------------------------------------
__global__ void k_h(const bf16_t* __restrict__ A, const bf16_t* __restrict__ Wt,
                    const float* __restrict__ emb, const int* __restrict__ Tsrc,
                    int tcol, int nrows,
                    float* __restrict__ hf, bf16_t* __restrict__ hb) {
    int wid  = (blockIdx.x * blockDim.x + threadIdx.x) >> 5;
    int lane = threadIdx.x & 31;
    int mtiles = nrows >> 4;
    const int nt64 = DIM / 64;
    if (wid >= mtiles * nt64) return;           // wave-uniform: EXEC stays full
    int mt = wid % mtiles, nt = wid / mtiles;
    int mbase = mt * 16, nbase = nt * 64;

    v8f acc[4] = {};
    gemm_16x64(A, Wt, mbase, nbase, lane, acc);

    int colb  = lane & 15;
    int rhalf = (lane >> 4) << 3;
    int tok[8];
    #pragma unroll
    for (int r = 0; r < 8; ++r) {
        int row = mbase + rhalf + r;
        tok[r] = Tsrc ? Tsrc[row * MAXLEN + tcol] : BOS_T;
    }
    #pragma unroll
    for (int j = 0; j < 4; ++j) {
        int col = nbase + j * 16 + colb;
        #pragma unroll
        for (int r = 0; r < 8; ++r) {
            int row = mbase + rhalf + r;
            float v = acc[j][r] + emb[(long long)tok[r] * DIM + col];
            v = tanhf(v);
            hf[(long long)row * DIM + col] = v;
            hb[(long long)row * DIM + col] = __float2bfloat16(v);
        }
    }
}

// ---------------------------------------------------------------------------
// raw = h @ w_out ; [nrows x VPAD], bf16 WMMA, wave = 16x64 tile.
// Consecutive waves in a block share one 64-col B strip (mt = wid % mtiles),
// so B re-reads hit the WGP cache and L2 traffic stays ~1x the bf16 weights.
// ---------------------------------------------------------------------------
__global__ void k_logits(const bf16_t* __restrict__ A, const bf16_t* __restrict__ Wt,
                         float* __restrict__ raw, int nrows) {
    int wid  = (blockIdx.x * blockDim.x + threadIdx.x) >> 5;
    int lane = threadIdx.x & 31;
    int mtiles = nrows >> 4;
    const int nt64 = VPAD / 64;                 // 786
    if (wid >= mtiles * nt64) return;
    int mt = wid % mtiles, nt = wid / mtiles;
    int mbase = mt * 16, nbase = nt * 64;

    v8f acc[4] = {};
    gemm_16x64(A, Wt, mbase, nbase, lane, acc);

    int colb  = lane & 15;
    int rhalf = (lane >> 4) << 3;
    #pragma unroll
    for (int j = 0; j < 4; ++j) {
        int col = nbase + j * 16 + colb;
        if (col < VOCAB) {
            #pragma unroll
            for (int r = 0; r < 8; ++r) {
                int row = mbase + rhalf + r;
                raw[(long long)row * VPAD + col] = acc[j][r];
            }
        }
    }
}

// ---------------------------------------------------------------------------
// Repetition penalty (gather-then-scatter, matches .at[idx].set semantics)
// plus forced EOS on the last step.  One thread per beam row.
// ---------------------------------------------------------------------------
__global__ void k_penalty(float* __restrict__ raw, const int* __restrict__ T, int t) {
    int n = threadIdx.x;
    if (n >= NROWS) return;
    const int* Tr = T + n * MAXLEN;
    float* R = raw + (long long)n * VPAD;
    int   toks[MAXLEN];
    float vals[MAXLEN];
    #pragma unroll
    for (int j = 0; j < MAXLEN; ++j)
        if (j < t) { toks[j] = Tr[j]; vals[j] = R[toks[j]]; }
    #pragma unroll
    for (int j = 0; j < MAXLEN; ++j)
        if (j < t) {
            float g = vals[j];
            R[toks[j]] = (g < 0.f) ? g * 1.2f : g * (1.0f / 1.2f);
        }
    if (t + 1 == MAXLEN) R[EOS_T] += 1e32f;
}

// ---------------------------------------------------------------------------
// Per-row constant c = max + log(sum exp) - beam_score, so that
// score[v] = raw[v] - c  ==  log_softmax + beam_score.
// ---------------------------------------------------------------------------
__global__ void k_rowstats(const float* __restrict__ raw, const float* __restrict__ bs,
                           float* __restrict__ rowconst) {
    int row = blockIdx.x, tid = threadIdx.x;
    __shared__ float red[256];
    const float* R = raw + (long long)row * VPAD;
    float m = NEG_BIG;
    for (int v = tid; v < VOCAB; v += 256) m = fmaxf(m, R[v]);
    red[tid] = m; __syncthreads();
    for (int o = 128; o > 0; o >>= 1) {
        if (tid < o) red[tid] = fmaxf(red[tid], red[tid + o]);
        __syncthreads();
    }
    float rm = red[0]; __syncthreads();
    float s = 0.f;
    for (int v = tid; v < VOCAB; v += 256) s += expf(R[v] - rm);
    red[tid] = s; __syncthreads();
    for (int o = 128; o > 0; o >>= 1) {
        if (tid < o) red[tid] += red[tid + o];
        __syncthreads();
    }
    if (tid == 0) rowconst[row] = rm + logf(red[0]) - (bs ? bs[row] : 0.f);
}

__global__ void k_scores(float* __restrict__ raw, const float* __restrict__ rowconst,
                         int nrows) {
    long long total = (long long)nrows * VOCAB;
    for (long long i = (long long)blockIdx.x * blockDim.x + threadIdx.x; i < total;
         i += (long long)gridDim.x * blockDim.x) {
        int row = (int)(i / VOCAB);
        int v   = (int)(i % VOCAB);
        raw[(long long)row * VPAD + v] -= rowconst[row];
    }
}

// ---------------------------------------------------------------------------
// Per-batch top-k over beams*VOCAB candidates, then keep first NBEAM non-EOS
// (topk order preserved == stable argsort on EOS flag).
// One block per batch row; per-thread sorted top-8 then LDS argmax rounds.
// ---------------------------------------------------------------------------
__global__ void k_topk(const float* __restrict__ sc, int beams, int topk,
                       int* __restrict__ kt, float* __restrict__ ks,
                       int* __restrict__ reorder) {
    int b = blockIdx.x, tid = threadIdx.x;
    __shared__ float Ls[2048]; __shared__ int Li[2048];
    __shared__ float rs[256];  __shared__ int rp[256];
    __shared__ float candS[8]; __shared__ int candI[8];

    float ls[8]; int li[8];
    #pragma unroll
    for (int q = 0; q < 8; ++q) { ls[q] = NEG_BIG; li[q] = 0; }

    for (int beam = 0; beam < beams; ++beam) {
        const float* R = sc + (long long)(b * beams + beam) * VPAD;
        for (int v = tid; v < VOCAB; v += 256) {
            float s = R[v];
            if (s > ls[7]) {
                float cs = s; int ci = beam * VOCAB + v;
                #pragma unroll
                for (int q = 0; q < 8; ++q) {
                    if (cs > ls[q]) {
                        float t0 = ls[q]; int t1 = li[q];
                        ls[q] = cs; li[q] = ci; cs = t0; ci = t1;
                    }
                }
            }
        }
    }
    #pragma unroll
    for (int q = 0; q < 8; ++q) { Ls[tid * 8 + q] = ls[q]; Li[tid * 8 + q] = li[q]; }
    __syncthreads();

    for (int r = 0; r < topk; ++r) {
        float bv = NEG_BIG; int bp = 0;
        #pragma unroll
        for (int q = 0; q < 8; ++q) {
            float v = Ls[tid * 8 + q];
            if (v > bv) { bv = v; bp = tid * 8 + q; }
        }
        rs[tid] = bv; rp[tid] = bp; __syncthreads();
        for (int o = 128; o > 0; o >>= 1) {
            if (tid < o && rs[tid + o] > rs[tid]) { rs[tid] = rs[tid + o]; rp[tid] = rp[tid + o]; }
            __syncthreads();
        }
        if (tid == 0) {
            int p = rp[0];
            candS[r] = rs[0]; candI[r] = Li[p];
            Ls[p] = NEG_BIG;
        }
        __syncthreads();
    }

    if (tid == 0) {
        int cnt = 0;
        for (int r = 0; r < topk && cnt < NBEAM; ++r) {
            int id  = candI[r];
            int tok = id % VOCAB;
            int fb  = id / VOCAB;
            if (tok != EOS_T) {
                kt[b * NBEAM + cnt] = tok;
                ks[b * NBEAM + cnt] = candS[r];
                if (reorder) reorder[b * NBEAM + cnt] = b * beams + fb;
                ++cnt;
            }
        }
    }
}

// ---------------------------------------------------------------------------
// First-step expansion: replicate h across beams, seed token buffer/scores.
// ---------------------------------------------------------------------------
__global__ void k_init(const float* __restrict__ htmp, bf16_t* __restrict__ hstate,
                       int* __restrict__ T0, float* __restrict__ bscore,
                       const int* __restrict__ kt, const float* __restrict__ ks) {
    int i = blockIdx.x;            // 0..127
    int b = i >> 2;
    int tid = threadIdx.x;
    for (int c = tid; c < DIM; c += blockDim.x)
        hstate[(long long)i * DIM + c] = __float2bfloat16(htmp[(long long)b * DIM + c]);
    if (tid < MAXLEN) {
        int v = PAD_T;
        if (tid == 0) v = BOS_T; else if (tid == 1) v = kt[i];
        T0[i * MAXLEN + tid] = v;
    }
    if (tid == 0) bscore[i] = ks[i];
}

// ---------------------------------------------------------------------------
// Beam reorder + token append + score update.
// ---------------------------------------------------------------------------
__global__ void k_reorder(const float* __restrict__ htmp, bf16_t* __restrict__ hstate,
                          const int* __restrict__ Tsrc, int* __restrict__ Tdst,
                          const int* __restrict__ kt, const float* __restrict__ ks,
                          const int* __restrict__ reorder, float* __restrict__ bscore,
                          int t) {
    int i = blockIdx.x, tid = threadIdx.x;
    int src = reorder[i];
    for (int c = tid; c < DIM; c += blockDim.x)
        hstate[(long long)i * DIM + c] = __float2bfloat16(htmp[(long long)src * DIM + c]);
    if (tid < MAXLEN) {
        int v = Tsrc[src * MAXLEN + tid];
        if (tid == t) v = kt[i];
        Tdst[i * MAXLEN + tid] = v;
    }
    if (tid == 0) bscore[i] = ks[i];
}

__global__ void k_out(const int* __restrict__ Tfin, const float* __restrict__ bscore,
                      float* __restrict__ out) {
    int b = blockIdx.x, tid = threadIdx.x;
    if (tid < MAXLEN) out[b * MAXLEN + tid] = (float)Tfin[(b * NBEAM) * MAXLEN + tid];
    if (tid < NBEAM)  out[BATCH * MAXLEN + b * NBEAM + tid] = bscore[b * NBEAM + tid];
}

// ===========================================================================
extern "C" void kernel_launch(void* const* d_in, const int* in_sizes, int n_in,
                              void* d_out, int out_size, void* d_ws, size_t ws_size,
                              hipStream_t stream) {
    const float* emb   = (const float*)d_in[0];
    const float* w_h   = (const float*)d_in[1];
    const float* w_out = (const float*)d_in[2];
    const float* h0    = (const float*)d_in[3];
    (void)in_sizes; (void)n_in; (void)out_size; (void)ws_size;

    char* ws = (char*)d_ws;
    size_t off = 0;
    auto take = [&](size_t bytes) {
        char* p = ws + off;
        off = (off + bytes + 255) & ~(size_t)255;
        return p;
    };
    bf16_t* wht    = (bf16_t*)take((size_t)DIM * DIM * 2);
    bf16_t* wot    = (bf16_t*)take((size_t)VPAD * DIM * 2);
    bf16_t* h0b    = (bf16_t*)take((size_t)BATCH * DIM * 2);
    bf16_t* hstate = (bf16_t*)take((size_t)NROWS * DIM * 2);
    float*  htmpf  = (float*) take((size_t)NROWS * DIM * 4);
    bf16_t* htmpb  = (bf16_t*)take((size_t)NROWS * DIM * 2);
    float*  raw    = (float*) take((size_t)NROWS * VPAD * 4);
    int*    Tbuf0  = (int*)   take((size_t)NROWS * MAXLEN * 4);
    int*    Tbuf1  = (int*)   take((size_t)NROWS * MAXLEN * 4);
    float*  bscore = (float*) take((size_t)NROWS * 4);
    float*  rowc   = (float*) take((size_t)NROWS * 4);
    int*    kt     = (int*)   take((size_t)BATCH * NBEAM * 4);
    float*  ks     = (float*) take((size_t)BATCH * NBEAM * 4);
    int*    reord  = (int*)   take((size_t)NROWS * 4);

    auto cdiv = [](long long a, long long b) { return (int)((a + b - 1) / b); };

    // ---- one-time weight prep (runs every call: deterministic) ----
    k_conv_T<<<8192, 256, 0, stream>>>(w_out, wot, DIM, VOCAB, VPAD);
    k_conv_T<<<4096, 256, 0, stream>>>(w_h, wht, DIM, DIM, DIM);
    k_conv  <<<cdiv(BATCH * DIM, 256), 256, 0, stream>>>(h0, h0b, BATCH * DIM);

    // ---- first step: decode BOS on 32 rows, top-5, keep 4 non-EOS ----
    {
        int waves = (BATCH / 16) * (DIM / 64);
        k_h<<<cdiv((long long)waves * 32, 256), 256, 0, stream>>>(
            h0b, wht, emb, (const int*)nullptr, 0, BATCH, htmpf, htmpb);
        int lw = (BATCH / 16) * (VPAD / 64);
        k_logits<<<cdiv((long long)lw * 32, 256), 256, 0, stream>>>(
            htmpb, wot, raw, BATCH);
        k_rowstats<<<BATCH, 256, 0, stream>>>(raw, (const float*)nullptr, rowc);
        k_scores<<<cdiv((long long)BATCH * VOCAB, 256), 256, 0, stream>>>(raw, rowc, BATCH);
        k_topk<<<BATCH, 256, 0, stream>>>(raw, 1, NBEAM + 1, kt, ks, (int*)nullptr);
        k_init<<<NROWS, 256, 0, stream>>>(htmpf, hstate, Tbuf0, bscore, kt, ks);
    }

    // ---- steps t = 2 .. 31 ----
    int hw = (NROWS / 16) * (DIM / 64);
    int lw = (NROWS / 16) * (VPAD / 64);
    for (int t = 2; t < MAXLEN; ++t) {
        int* Tsrc = (t & 1) ? Tbuf1 : Tbuf0;
        int* Tdst = (t & 1) ? Tbuf0 : Tbuf1;
        k_h<<<cdiv((long long)hw * 32, 256), 256, 0, stream>>>(
            hstate, wht, emb, Tsrc, t - 1, NROWS, htmpf, htmpb);
        k_logits<<<cdiv((long long)lw * 32, 256), 256, 0, stream>>>(
            htmpb, wot, raw, NROWS);
        k_penalty<<<1, NROWS, 0, stream>>>(raw, Tsrc, t);
        k_rowstats<<<NROWS, 256, 0, stream>>>(raw, bscore, rowc);
        k_scores<<<cdiv((long long)NROWS * VOCAB, 256), 256, 0, stream>>>(raw, rowc, NROWS);
        k_topk<<<BATCH, 256, 0, stream>>>(raw, NBEAM, 2 * NBEAM, kt, ks, reord);
        k_reorder<<<NROWS, 256, 0, stream>>>(htmpf, hstate, Tsrc, Tdst, kt, ks,
                                             reord, bscore, t);
    }

    // final state is in Tbuf0 (last t = 31 is odd -> dst = Tbuf0)
    k_out<<<BATCH, 64, 0, stream>>>(Tbuf0, bscore, (float*)d_out);
}
